// MeshGraphNet_26903675142327
// MI455X (gfx1250) — matrix operand order repaired and checked by reading, not compiled, not run on hardware
//
#include <hip/hip_runtime.h>

#define NN 10000
#define NE 160000
#define LAT 128
#define LN_EPS 1e-5f

typedef __attribute__((ext_vector_type(16))) _Float16 v16h;
typedef __attribute__((ext_vector_type(8)))  _Float16 v8h;
typedef __attribute__((ext_vector_type(4)))  _Float16 v4h;
typedef __attribute__((ext_vector_type(8)))  float    v8f;
typedef __attribute__((ext_vector_type(4)))  float    v4f;

union V16H { v16h v; v8h h[2]; };

// ---------------------------------------------------------------- WMMA core

__device__ __forceinline__ v8f wmma16(v16h a, v16h b, v8f c) {
  return __builtin_amdgcn_wmma_f32_16x16x32_f16(false, a, false, b, (short)0, c,
                                                false, false);
}

// A fragment from precomputed per-lane base: two contiguous 16B runs (2x ds_read_b128)
__device__ __forceinline__ v16h a_frag(const _Float16* p) {
  V16H u;
  u.h[0] = *(const v8h*)p;
  u.h[1] = *(const v8h*)(p + 16);
  return u.v;
}

__device__ __forceinline__ void clear_acc(v8f acc[4]) {
#pragma unroll
  for (int i = 0; i < 4; ++i)
#pragma unroll
    for (int j = 0; j < 8; ++j) acc[i][j] = 0.0f;
}

// 64 rows x K (LDS f16) @ W[K x 128] (packed f16) -> this wave's 64x16 slab.
// Software-pipelined (depth 2): iteration kk issues loads for kk+1 before the
// 4 WMMAs of kk, hiding LDS/global latency under matrix-op execution.
template <int KSTEPS>
__device__ __forceinline__ void mlp_layer(const _Float16* ldsA, int pitch,
                                          const _Float16* wpk, int lane, int wave,
                                          v8f acc[4]) {
  const _Float16* aBase = ldsA + (lane & 15) * pitch + (lane >> 4) * 8;
  const _Float16* bBase = wpk + (size_t)wave * 512 + lane * 16;
  v16h b  = *(const v16h*)(bBase);
  v16h a0 = a_frag(aBase);
  v16h a1 = a_frag(aBase + 16 * pitch);
  v16h a2 = a_frag(aBase + 32 * pitch);
  v16h a3 = a_frag(aBase + 48 * pitch);
#pragma unroll
  for (int kk = 0; kk < KSTEPS; ++kk) {
    v16h bn = b, an0 = a0, an1 = a1, an2 = a2, an3 = a3;
    if (kk + 1 < KSTEPS) {
      bn  = *(const v16h*)(bBase + (size_t)(kk + 1) * 8 * 512);
      an0 = a_frag(aBase + (kk + 1) * 32);
      an1 = a_frag(aBase + 16 * pitch + (kk + 1) * 32);
      an2 = a_frag(aBase + 32 * pitch + (kk + 1) * 32);
      an3 = a_frag(aBase + 48 * pitch + (kk + 1) * 32);
    }
    acc[0] = wmma16(a0, b, acc[0]);
    acc[1] = wmma16(a1, b, acc[1]);
    acc[2] = wmma16(a2, b, acc[2]);
    acc[3] = wmma16(a3, b, acc[3]);
    b = bn; a0 = an0; a1 = an1; a2 = an2; a3 = an3;
  }
}

// C/D layout: lane group g holds rows g*8+r, col = lane&15 (per 16x16 subtile).
__device__ __forceinline__ void relu_store_h(const v8f acc[4], const float* bias,
                                             _Float16* ldsOut, int pitchO,
                                             int lane, int wave) {
  int n = lane & 15, g = lane >> 4;
  int col = wave * 16 + n;
  float bb = bias[col];
#pragma unroll
  for (int s = 0; s < 4; ++s)
#pragma unroll
    for (int r = 0; r < 8; ++r) {
      float x = acc[s][r] + bb;
      ldsOut[(s * 16 + g * 8 + r) * pitchO + col] = (_Float16)(x > 0.0f ? x : 0.0f);
    }
}

#define FP 132  // f32 LDS pitch (132 % 64 == 4 -> conflict-free row access)
__device__ __forceinline__ void store_f32_lds(const v8f acc[4], const float* bias,
                                              float* F, int lane, int wave) {
  int n = lane & 15, g = lane >> 4;
  int col = wave * 16 + n;
  float bb = bias[col];
#pragma unroll
  for (int s = 0; s < 4; ++s)
#pragma unroll
    for (int r = 0; r < 8; ++r)
      F[(s * 16 + g * 8 + r) * FP + col] = acc[s][r] + bb;
}

// Parallel LayerNorm stats: 256 threads make 64x4 partials, 64 threads combine.
__device__ __forceinline__ void ln_stats(const float* F, float* s_ps, float* s_pq,
                                         float* s_mu, float* s_rs, int t) {
  int row = t & 63, qi = t >> 6;
  const float* fr = F + row * FP + qi * 32;
  float sm = 0.0f, sq = 0.0f;
#pragma unroll
  for (int c = 0; c < 32; ++c) { float x = fr[c]; sm += x; sq += x * x; }
  s_ps[qi * 64 + row] = sm;
  s_pq[qi * 64 + row] = sq;
  __syncthreads();
  if (t < 64) {
    float m = (s_ps[t] + s_ps[64 + t] + s_ps[128 + t] + s_ps[192 + t]) * (1.0f / 128.0f);
    float q = (s_pq[t] + s_pq[64 + t] + s_pq[128 + t] + s_pq[192 + t]) * (1.0f / 128.0f);
    s_mu[t] = m;
    s_rs[t] = rsqrtf(q - m * m + LN_EPS);
  }
  __syncthreads();
}

// ------------------------------------------------- async global->LDS staging
// CDNA5 async copy: per-lane 16B global -> LDS, tracked by ASYNCcnt.
__device__ __forceinline__ void async_g2l_b128(unsigned lds_off, const void* gptr) {
  asm volatile("global_load_async_to_lds_b128 %0, %1, off"
               :
               : "v"(lds_off), "v"((unsigned long long)(uintptr_t)gptr)
               : "memory");
}
__device__ __forceinline__ void wait_async0() {
  asm volatile("s_wait_asynccnt 0" ::: "memory");
}

// ---------------------------------------------------------------- weight pack
// f32 W[KxN] row-major -> f16 WMMA-B fragment blocks (512 halfs per block).
__global__ void pack_w_kernel(const float* __restrict__ w, _Float16* __restrict__ dst,
                              int K, int N, int nkT, int nnT) {
  int total = nkT * nnT * 512;
  for (int i = blockIdx.x * blockDim.x + threadIdx.x; i < total;
       i += gridDim.x * blockDim.x) {
    int j = i & 15;
    int lane = (i >> 4) & 31;
    int blk = i >> 9;
    int nt = blk % nnT, kk = blk / nnT;
    int k = kk * 32 + (lane >> 4) * 8 + (j & 7) + (j >> 3) * 16;
    int n = nt * 16 + (lane & 15);
    float val = (k < K && n < N) ? w[(size_t)k * N + n] : 0.0f;
    dst[i] = (_Float16)val;
  }
}

__global__ void zero_f32_kernel(float* __restrict__ p, int n) {
  int i = blockIdx.x * blockDim.x + threadIdx.x;
  if (i < n) p[i] = 0.0f;
}

// ---------------------------------------------------------------- encoders
#define AP0 40   // (32+8) halfs
#define HP  136  // (128+8) halfs

__global__ __launch_bounds__(256) void encode_kernel(
    const float* __restrict__ feat, int fin, int nRows,
    const _Float16* __restrict__ w0, const float* __restrict__ b0,
    const _Float16* __restrict__ w1, const float* __restrict__ b1,
    const _Float16* __restrict__ w2, const float* __restrict__ b2,
    const float* __restrict__ gamma, const float* __restrict__ beta,
    float* __restrict__ out, _Float16* __restrict__ out_h) {
  __shared__ __align__(16) unsigned char smem[5120 + 17408 + 17408 + 33792];
  __shared__ float s_mu[64], s_rs[64], s_ps[256], s_pq[256];
  _Float16* A0 = (_Float16*)smem;
  _Float16* H1 = (_Float16*)(smem + 5120);
  _Float16* H2 = (_Float16*)(smem + 22528);
  float*    F  = (float*)(smem + 39936);
  const int t = threadIdx.x, lane = t & 31, wave = t >> 5;
  const int rowBase = blockIdx.x * 64;

  if (t < 64) {
    int gr = rowBase + t;
    for (int k = 0; k < 32; ++k) {
      float val = (k < fin && gr < nRows) ? feat[(size_t)gr * fin + k] : 0.0f;
      A0[t * AP0 + k] = (_Float16)val;
    }
  }
  __syncthreads();

  v8f acc[4];
  clear_acc(acc);
  mlp_layer<1>(A0, AP0, w0, lane, wave, acc);
  relu_store_h(acc, b0, H1, HP, lane, wave);
  __syncthreads();
  clear_acc(acc);
  mlp_layer<4>(H1, HP, w1, lane, wave, acc);
  relu_store_h(acc, b1, H2, HP, lane, wave);
  __syncthreads();
  clear_acc(acc);
  mlp_layer<4>(H2, HP, w2, lane, wave, acc);
  store_f32_lds(acc, b2, F, lane, wave);
  __syncthreads();

  ln_stats(F, s_ps, s_pq, s_mu, s_rs, t);
  {
    int row = t & 63, c0 = (t >> 6) * 32;
    int gr = rowBase + row;
    if (gr < nRows) {
      float mu = s_mu[row], rs = s_rs[row];
      const float* fr = F + row * FP;
      float*     orow = out + (size_t)gr * LAT;
      _Float16*  hrow = out_h + (size_t)gr * LAT;
      for (int c = c0; c < c0 + 32; c += 4) {
        v4f x  = *(const v4f*)(fr + c);
        v4f gm = *(const v4f*)(gamma + c);
        v4f bt = *(const v4f*)(beta + c);
        v4f r;
        r.x = (x.x - mu) * rs * gm.x + bt.x;
        r.y = (x.y - mu) * rs * gm.y + bt.y;
        r.z = (x.z - mu) * rs * gm.z + bt.z;
        r.w = (x.w - mu) * rs * gm.w + bt.w;
        *(v4f*)(orow + c) = r;
        v4h hh;
        hh.x = (_Float16)r.x; hh.y = (_Float16)r.y;
        hh.z = (_Float16)r.z; hh.w = (_Float16)r.w;
        *(v4h*)(hrow + c) = hh;
      }
    }
  }
}

// ---------------------------------------------------------------- edge step
// e += LN(MLP([e, v[src], v[dst]])) ; agg[dst] += e ; maintain f16 mirror e_h
#define EP0 392  // (384+8) halfs

__global__ __launch_bounds__(256) void edge_step_kernel(
    float* __restrict__ e, _Float16* __restrict__ e_h,
    const _Float16* __restrict__ v_h,
    const int* __restrict__ srcI, const int* __restrict__ dstI,
    const _Float16* __restrict__ w0, const float* __restrict__ b0,
    const _Float16* __restrict__ w1, const float* __restrict__ b1,
    const _Float16* __restrict__ w2, const float* __restrict__ b2,
    const float* __restrict__ gamma, const float* __restrict__ beta,
    float* __restrict__ agg) {
  __shared__ __align__(16) unsigned char smem[50176 + 17408 + 17408];
  __shared__ float s_mu[64], s_rs[64], s_ps[256], s_pq[256];
  _Float16* A0 = (_Float16*)smem;
  _Float16* H1 = (_Float16*)(smem + 50176);
  _Float16* H2 = (_Float16*)(smem + 67584);
  float*    F  = (float*)smem;  // overlays A0 after layer1
  const int t = threadIdx.x, lane = t & 31, wave = t >> 5;
  const int rowBase = blockIdx.x * 64;

  // async-stage concat(e_h, v_h[src], v_h[dst]) -> LDS; 48 x b128 chunks per row
  for (int q = t; q < 64 * 48; q += 256) {
    int row = q / 48, ch = q - row * 48;
    int col = ch * 8;
    int gr = rowBase + row;
    const _Float16* gp;
    if (col < 128)      gp = e_h + (size_t)gr * LAT + col;
    else if (col < 256) gp = v_h + (size_t)srcI[gr] * LAT + (col - 128);
    else                gp = v_h + (size_t)dstI[gr] * LAT + (col - 256);
    async_g2l_b128((unsigned)(uintptr_t)(A0 + row * EP0 + col), gp);
  }
  wait_async0();
  __syncthreads();

  v8f acc[4];
  clear_acc(acc);
  mlp_layer<12>(A0, EP0, w0, lane, wave, acc);  // 384 -> 128
  relu_store_h(acc, b0, H1, HP, lane, wave);
  __syncthreads();
  clear_acc(acc);
  mlp_layer<4>(H1, HP, w1, lane, wave, acc);    // 128 -> 128
  relu_store_h(acc, b1, H2, HP, lane, wave);
  __syncthreads();
  clear_acc(acc);
  mlp_layer<4>(H2, HP, w2, lane, wave, acc);    // 128 -> 128
  store_f32_lds(acc, b2, F, lane, wave);
  __syncthreads();

  ln_stats(F, s_ps, s_pq, s_mu, s_rs, t);
  {
    int row = t & 63, c0 = (t >> 6) * 32;
    int gr = rowBase + row;
    float mu = s_mu[row], rs = s_rs[row];
    int dn = dstI[gr];
    float*    erow = e + (size_t)gr * LAT;
    _Float16* hrow = e_h + (size_t)gr * LAT;
    float*    arow = agg + (size_t)dn * LAT;
    const float* fr = F + row * FP;
    for (int c = c0; c < c0 + 32; c += 4) {
      v4f x  = *(const v4f*)(fr + c);
      v4f gm = *(const v4f*)(gamma + c);
      v4f bt = *(const v4f*)(beta + c);
      v4f eo = *(const v4f*)(erow + c);
      v4f r;
      r.x = eo.x + (x.x - mu) * rs * gm.x + bt.x;
      r.y = eo.y + (x.y - mu) * rs * gm.y + bt.y;
      r.z = eo.z + (x.z - mu) * rs * gm.z + bt.z;
      r.w = eo.w + (x.w - mu) * rs * gm.w + bt.w;
      *(v4f*)(erow + c) = r;
      v4h hh;
      hh.x = (_Float16)r.x; hh.y = (_Float16)r.y;
      hh.z = (_Float16)r.z; hh.w = (_Float16)r.w;
      *(v4h*)(hrow + c) = hh;
      atomicAdd(arow + c + 0, r.x);
      atomicAdd(arow + c + 1, r.y);
      atomicAdd(arow + c + 2, r.z);
      atomicAdd(arow + c + 3, r.w);
    }
  }
}

// ---------------------------------------------------------------- node step
// v += LN(MLP([v, agg])) ; maintain f16 mirror v_h
#define NP0 264  // (256+8) halfs

__global__ __launch_bounds__(256) void node_step_kernel(
    float* __restrict__ v, _Float16* __restrict__ v_h,
    const float* __restrict__ agg,
    const _Float16* __restrict__ w0, const float* __restrict__ b0,
    const _Float16* __restrict__ w1, const float* __restrict__ b1,
    const _Float16* __restrict__ w2, const float* __restrict__ b2,
    const float* __restrict__ gamma, const float* __restrict__ beta, int nRows) {
  __shared__ __align__(16) unsigned char smem[33792 + 17408 + 17408];
  __shared__ float s_mu[64], s_rs[64], s_ps[256], s_pq[256];
  _Float16* A0 = (_Float16*)smem;
  _Float16* H1 = (_Float16*)(smem + 33792);
  _Float16* H2 = (_Float16*)(smem + 51200);
  float*    F  = (float*)smem;  // overlays A0 after layer1
  const int t = threadIdx.x, lane = t & 31, wave = t >> 5;
  const int rowBase = blockIdx.x * 64;

  // v half: async f16 rows (rows past nRows read in-bounds scratch garbage; unused)
  for (int q = t; q < 64 * 16; q += 256) {
    int row = q >> 4, ch = q & 15;
    int gr = rowBase + row;
    async_g2l_b128((unsigned)(uintptr_t)(A0 + row * NP0 + ch * 8),
                   v_h + (size_t)gr * LAT + ch * 8);
  }
  // agg half: f32 -> f16 convert
  for (int q = t; q < 64 * 32; q += 256) {
    int row = q >> 5, col = 128 + (q & 31) * 4;
    int gr = rowBase + row;
    v4f x;
    if (gr < nRows) x = *(const v4f*)(agg + (size_t)gr * LAT + (col - 128));
    else { x.x = 0.0f; x.y = 0.0f; x.z = 0.0f; x.w = 0.0f; }
    v4h h;
    h.x = (_Float16)x.x; h.y = (_Float16)x.y;
    h.z = (_Float16)x.z; h.w = (_Float16)x.w;
    *(v4h*)(A0 + row * NP0 + col) = h;
  }
  wait_async0();
  __syncthreads();

  v8f acc[4];
  clear_acc(acc);
  mlp_layer<8>(A0, NP0, w0, lane, wave, acc);   // 256 -> 128
  relu_store_h(acc, b0, H1, HP, lane, wave);
  __syncthreads();
  clear_acc(acc);
  mlp_layer<4>(H1, HP, w1, lane, wave, acc);
  relu_store_h(acc, b1, H2, HP, lane, wave);
  __syncthreads();
  clear_acc(acc);
  mlp_layer<4>(H2, HP, w2, lane, wave, acc);
  store_f32_lds(acc, b2, F, lane, wave);
  __syncthreads();

  ln_stats(F, s_ps, s_pq, s_mu, s_rs, t);
  {
    int row = t & 63, c0 = (t >> 6) * 32;
    int gr = rowBase + row;
    if (gr < nRows) {
      float mu = s_mu[row], rs = s_rs[row];
      float*    vrow = v + (size_t)gr * LAT;
      _Float16* hrow = v_h + (size_t)gr * LAT;
      const float* fr = F + row * FP;
      for (int c = c0; c < c0 + 32; c += 4) {
        v4f x  = *(const v4f*)(fr + c);
        v4f gm = *(const v4f*)(gamma + c);
        v4f bt = *(const v4f*)(beta + c);
        v4f vo = *(const v4f*)(vrow + c);
        v4f r;
        r.x = vo.x + (x.x - mu) * rs * gm.x + bt.x;
        r.y = vo.y + (x.y - mu) * rs * gm.y + bt.y;
        r.z = vo.z + (x.z - mu) * rs * gm.z + bt.z;
        r.w = vo.w + (x.w - mu) * rs * gm.w + bt.w;
        *(v4f*)(vrow + c) = r;
        v4h hh;
        hh.x = (_Float16)r.x; hh.y = (_Float16)r.y;
        hh.z = (_Float16)r.z; hh.w = (_Float16)r.w;
        *(v4h*)(hrow + c) = hh;
      }
    }
  }
}

// ---------------------------------------------------------------- decoder
__global__ __launch_bounds__(256) void decode_kernel(
    const _Float16* __restrict__ v_h,
    const _Float16* __restrict__ w0, const float* __restrict__ b0,
    const _Float16* __restrict__ w1, const float* __restrict__ b1,
    const _Float16* __restrict__ w2, const float* __restrict__ b2,
    float* __restrict__ out, int nRows) {
  __shared__ __align__(16) unsigned char smem[17408 * 3];
  _Float16* A0 = (_Float16*)smem;
  _Float16* H1 = (_Float16*)(smem + 17408);
  _Float16* H2 = (_Float16*)(smem + 34816);
  const int t = threadIdx.x, lane = t & 31, wave = t >> 5;
  const int rowBase = blockIdx.x * 64;

  for (int q = t; q < 64 * 16; q += 256) {
    int row = q >> 4, ch = q & 15;
    int gr = rowBase + row;
    async_g2l_b128((unsigned)(uintptr_t)(A0 + row * HP + ch * 8),
                   v_h + (size_t)gr * LAT + ch * 8);
  }
  wait_async0();
  __syncthreads();

  v8f acc[4];
  clear_acc(acc);
  mlp_layer<4>(A0, HP, w0, lane, wave, acc);
  relu_store_h(acc, b0, H1, HP, lane, wave);
  __syncthreads();
  clear_acc(acc);
  mlp_layer<4>(H1, HP, w1, lane, wave, acc);
  relu_store_h(acc, b1, H2, HP, lane, wave);
  __syncthreads();

  if (wave == 0) {  // final 128 -> 3 (one 16-wide N tile, zero padded)
    clear_acc(acc);
    const _Float16* aBase = H2 + (lane & 15) * HP + (lane >> 4) * 8;
    const _Float16* bBase = w2 + lane * 16;
#pragma unroll
    for (int kk = 0; kk < 4; ++kk) {
      v16h b  = *(const v16h*)(bBase + (size_t)kk * 512);
      v16h a0 = a_frag(aBase + kk * 32);
      v16h a1 = a_frag(aBase + 16 * HP + kk * 32);
      v16h a2 = a_frag(aBase + 32 * HP + kk * 32);
      v16h a3 = a_frag(aBase + 48 * HP + kk * 32);
      acc[0] = wmma16(a0, b, acc[0]);
      acc[1] = wmma16(a1, b, acc[1]);
      acc[2] = wmma16(a2, b, acc[2]);
      acc[3] = wmma16(a3, b, acc[3]);
    }
    int n = lane & 15, g = lane >> 4;
    if (n < 3) {
      float bb = b2[n];
#pragma unroll
      for (int s = 0; s < 4; ++s)
#pragma unroll
        for (int r = 0; r < 8; ++r) {
          int gr = rowBase + s * 16 + g * 8 + r;
          if (gr < nRows) out[(size_t)gr * 3 + n] = acc[s][r] + bb;
        }
    }
  }
}

// ---------------------------------------------------------------- launch
static inline size_t pack_halfs(int K, int N) {
  return (size_t)((K + 31) / 32) * (size_t)((N + 15) / 16) * 512;
}

extern "C" void kernel_launch(void* const* d_in, const int* in_sizes, int n_in,
                              void* d_out, int out_size, void* d_ws, size_t ws_size,
                              hipStream_t stream) {
  (void)in_sizes; (void)n_in; (void)out_size; (void)ws_size;
  const float* node_f = (const float*)d_in[0];
  const float* edge_f = (const float*)d_in[1];
  const int*   eidx   = (const int*)d_in[2];
  const int* srcI = eidx;
  const int* dstI = eidx + NE;

  // param leaves in jax pytree (dict keys sorted) order:
  // dec(6) | enc_edge(6) | enc_edge_ln(2) | enc_node(6) | enc_node_ln(2) | 8 x steps(16)
  auto PF = [&](int i) { return (const float*)d_in[i]; };

  char* cur = (char*)d_ws;
  float*    v   = (float*)cur;    cur += (size_t)NN * LAT * 4;
  float*    e   = (float*)cur;    cur += (size_t)NE * LAT * 4;
  float*    agg = (float*)cur;    cur += (size_t)NN * LAT * 4;
  _Float16* v_h = (_Float16*)cur; cur += (size_t)NN * LAT * 2;
  _Float16* e_h = (_Float16*)cur; cur += (size_t)NE * LAT * 2;

  auto alloc_pack = [&](int K, int N) {
    _Float16* p = (_Float16*)cur;
    cur += pack_halfs(K, N) * 2;
    return p;
  };
  auto pack = [&](const float* src, _Float16* dst, int K, int N) {
    int nkT = (K + 31) / 32, nnT = (N + 15) / 16;
    int total = nkT * nnT * 512;
    pack_w_kernel<<<(total + 255) / 256, 256, 0, stream>>>(src, dst, K, N, nkT, nnT);
  };

  _Float16* decW[3] = {alloc_pack(128, 128), alloc_pack(128, 128), alloc_pack(128, 3)};
  pack(PF(3), decW[0], 128, 128);
  pack(PF(5), decW[1], 128, 128);
  pack(PF(7), decW[2], 128, 3);
  _Float16* eeW[3] = {alloc_pack(4, 128), alloc_pack(128, 128), alloc_pack(128, 128)};
  pack(PF(9),  eeW[0], 4, 128);
  pack(PF(11), eeW[1], 128, 128);
  pack(PF(13), eeW[2], 128, 128);
  _Float16* enW[3] = {alloc_pack(7, 128), alloc_pack(128, 128), alloc_pack(128, 128)};
  pack(PF(17), enW[0], 7, 128);
  pack(PF(19), enW[1], 128, 128);
  pack(PF(21), enW[2], 128, 128);
  _Float16* sEW[8][3];
  _Float16* sNW[8][3];
  for (int s = 0; s < 8; ++s) {
    int B = 25 + s * 16;
    sEW[s][0] = alloc_pack(384, 128); pack(PF(B + 2),  sEW[s][0], 384, 128);
    sEW[s][1] = alloc_pack(128, 128); pack(PF(B + 4),  sEW[s][1], 128, 128);
    sEW[s][2] = alloc_pack(128, 128); pack(PF(B + 6),  sEW[s][2], 128, 128);
    sNW[s][0] = alloc_pack(256, 128); pack(PF(B + 10), sNW[s][0], 256, 128);
    sNW[s][1] = alloc_pack(128, 128); pack(PF(B + 12), sNW[s][1], 128, 128);
    sNW[s][2] = alloc_pack(128, 128); pack(PF(B + 14), sNW[s][2], 128, 128);
  }

  const int nodeBlocks = (NN + 63) / 64;   // 157
  const int edgeBlocks = NE / 64;          // 2500

  encode_kernel<<<nodeBlocks, 256, 0, stream>>>(node_f, 7, NN, enW[0], PF(18),
                                                enW[1], PF(20), enW[2], PF(22),
                                                PF(23), PF(24), v, v_h);
  encode_kernel<<<edgeBlocks, 256, 0, stream>>>(edge_f, 4, NE, eeW[0], PF(10),
                                                eeW[1], PF(12), eeW[2], PF(14),
                                                PF(15), PF(16), e, e_h);

  for (int s = 0; s < 8; ++s) {
    int B = 25 + s * 16;
    zero_f32_kernel<<<(NN * LAT + 255) / 256, 256, 0, stream>>>(agg, NN * LAT);
    edge_step_kernel<<<edgeBlocks, 256, 0, stream>>>(
        e, e_h, v_h, srcI, dstI, sEW[s][0], PF(B + 3), sEW[s][1], PF(B + 5),
        sEW[s][2], PF(B + 7), PF(B + 0), PF(B + 1), agg);
    node_step_kernel<<<nodeBlocks, 256, 0, stream>>>(
        v, v_h, agg, sNW[s][0], PF(B + 11), sNW[s][1], PF(B + 13), sNW[s][2],
        PF(B + 15), PF(B + 8), PF(B + 9), NN);
  }

  decode_kernel<<<nodeBlocks, 256, 0, stream>>>(v_h, decW[0], PF(4), decW[1], PF(6),
                                                decW[2], PF(8), (float*)d_out, NN);
}
